// ReBased_75746043232513
// MI455X (gfx1250) — matrix operand order repaired and checked by reading, not compiled, not run on hardware
//
#include <hip/hip_runtime.h>

#define B_  2
#define L_  2048
#define DM_ 1536
#define H_  12
#define FD_ 16
#define DV_ 128
#define D2_ 256
#define CH_ 128
#define NC_ 16
#define BL_ (B_ * L_)

typedef __attribute__((ext_vector_type(16))) __bf16 v16bf;
typedef __attribute__((ext_vector_type(8)))  __bf16 v8bf;
typedef __attribute__((ext_vector_type(8)))  float  v8f;
typedef __attribute__((ext_vector_type(4)))  float  v4f;
typedef __attribute__((ext_vector_type(4)))  unsigned int v4u;
typedef __attribute__((ext_vector_type(4)))  int    v4i;

__device__ __forceinline__ v8f zero_v8f() {
  v8f z = {0.f, 0.f, 0.f, 0.f, 0.f, 0.f, 0.f, 0.f};
  return z;
}

// Column index within its 32-block in WMMA A-fragment element
// order [0..31] (contiguous: lane half h holds k 16h..16h+15).
__host__ __device__ __forceinline__ int swz32(int c) {
  int lc = c & 31;
  int pos = lc;
  return (c & ~31) + pos;
}

#define WMMA_BF16(af, bfv, acc)                                                \
  (acc) = __builtin_amdgcn_wmma_f32_16x16x32_bf16(false, (af), false, (bfv),   \
                                                  (short)0, (acc), false, false)

// ---- CDNA5 async global->LDS copy (ASYNCcnt path), with safe fallback ------
#if defined(__gfx1250__) && __has_builtin(__builtin_amdgcn_global_load_async_to_lds_b128)
#define HAVE_ASYNC_LDS 1
#else
#define HAVE_ASYNC_LDS 0
#endif

__device__ __forceinline__ void copy16_g2l(void* lds, const void* g) {
#if HAVE_ASYNC_LDS
  __builtin_amdgcn_global_load_async_to_lds_b128(
      (__attribute__((address_space(1))) v4i*)g,
      (__attribute__((address_space(3))) v4i*)lds, 0, 0);
#else
  *(v4u*)lds = *(const v4u*)g;
#endif
}

__device__ __forceinline__ void wait_async_lds() {
#if HAVE_ASYNC_LDS
#if __has_builtin(__builtin_amdgcn_s_wait_asynccnt)
  __builtin_amdgcn_s_wait_asynccnt(0);
#else
  asm volatile("s_wait_asynccnt 0" ::: "memory");
#endif
#endif
}

// ---------------------------------------------------------------------------
// Weight convert + transpose: W (K x N, f32) -> WT (N x K, bf16)
// ---------------------------------------------------------------------------
__global__ void k_wt_bf16(const float* __restrict__ W, __bf16* __restrict__ WT,
                          int K, int N) {
  long idx = (long)blockIdx.x * blockDim.x + threadIdx.x;
  long total = (long)K * N;
  if (idx >= total) return;
  int k = (int)(idx % K);
  int n = (int)(idx / K);
  WT[idx] = (__bf16)W[(long)k * N + n];
}

// ---------------------------------------------------------------------------
// A-matrix convert + fragment-order swizzle: X (rows x K f32, row-major) ->
// Y bf16 with each 32-column block permuted so a lane's 16 A-fragment
// elements are one contiguous 32B load.
// ---------------------------------------------------------------------------
__global__ void k_a_swz_bf16(const float* __restrict__ X, __bf16* __restrict__ Y,
                             long total, int K) {
  long idx = (long)blockIdx.x * blockDim.x + threadIdx.x;
  if (idx >= total) return;
  int c = (int)(idx % K);
  long row = idx / K;
  Y[row * K + swz32(c)] = (__bf16)X[idx];
}

// ---------------------------------------------------------------------------
// GEMM: C (MxN f32) = A (MxK bf16, fragment-swizzled) * B (NxK bf16, i.e. B^T).
// 8 waves/block; tile 128(M) x 64(N); wave -> 16 rows x 4 accs.
// Inner loop: 5 contiguous 32B v16bf loads + 4 WMMAs; ping-pong double buffer.
// ---------------------------------------------------------------------------
__global__ __launch_bounds__(256) void k_gemm(const __bf16* __restrict__ A,
                                              const __bf16* __restrict__ BT,
                                              float* __restrict__ C,
                                              int M, int N, int K) {
  const int lane = threadIdx.x & 31;
  const int wv   = threadIdx.x >> 5;
  const int half = lane >> 4;
  const int l16  = lane & 15;
  const int m0 = blockIdx.x * 128 + wv * 16;
  const int n0 = blockIdx.y * 64;

  v8f acc[4];
#pragma unroll
  for (int t = 0; t < 4; ++t) acc[t] = zero_v8f();

  const __bf16* arow = A + (size_t)(m0 + l16) * K + half * 16;
  const __bf16* brow[4];
#pragma unroll
  for (int t = 0; t < 4; ++t)
    brow[t] = BT + (size_t)(n0 + t * 16 + l16) * K + half * 16;

  // prologue (k = 0)
  v16bf af0 = *(const v16bf*)(arow);
  v16bf bv0[4];
#pragma unroll
  for (int t = 0; t < 4; ++t) bv0[t] = *(const v16bf*)(brow[t]);

  for (int k = 0; k < K; k += 64) {
    const int k1 = k + 32;
    v16bf af1 = *(const v16bf*)(arow + k1);
    v16bf bv1[4];
#pragma unroll
    for (int t = 0; t < 4; ++t) bv1[t] = *(const v16bf*)(brow[t] + k1);
    __builtin_prefetch((const void*)(arow + k1 + 512), 0, 1);
#pragma unroll
    for (int t = 0; t < 4; ++t) WMMA_BF16(af0, bv0[t], acc[t]);

    const int k2 = (k + 64 < K) ? (k + 64) : 0;   // dummy reload on last iter
    af0 = *(const v16bf*)(arow + k2);
#pragma unroll
    for (int t = 0; t < 4; ++t) bv0[t] = *(const v16bf*)(brow[t] + k2);
#pragma unroll
    for (int t = 0; t < 4; ++t) WMMA_BF16(af1, bv1[t], acc[t]);
  }

#pragma unroll
  for (int t = 0; t < 4; ++t)
#pragma unroll
    for (int r = 0; r < 8; ++r)
      C[(size_t)(m0 + half * 8 + r) * N + (n0 + t * 16 + l16)] = acc[t][r];
}

// ---------------------------------------------------------------------------
// LayerNorm over 192 features + head-split + fold 0.5 (Taylor 1/sqrt(FD)
// split as 0.5 per factor). One wave per token row. Out: (B, H, L, FD) bf16.
// ---------------------------------------------------------------------------
__global__ __launch_bounds__(256) void k_ln_split(const float* __restrict__ X,
                                                  const float* __restrict__ g,
                                                  const float* __restrict__ bta,
                                                  __bf16* __restrict__ Outh) {
  int row  = blockIdx.x * 8 + (threadIdx.x >> 5);
  int lane = threadIdx.x & 31;
  const float* x = X + (size_t)row * (H_ * FD_);
  float v[6];
  float s = 0.f;
#pragma unroll
  for (int i = 0; i < 6; ++i) { v[i] = x[lane + 32 * i]; s += v[i]; }
#pragma unroll
  for (int off = 16; off; off >>= 1) s += __shfl_xor(s, off, 32);
  float mu = s * (1.f / 192.f);
  float s2 = 0.f;
#pragma unroll
  for (int i = 0; i < 6; ++i) { float d = v[i] - mu; s2 += d * d; }
#pragma unroll
  for (int off = 16; off; off >>= 1) s2 += __shfl_xor(s2, off, 32);
  float rs = rsqrtf(s2 * (1.f / 192.f) + 1e-5f);
  int b = row >> 11;
  int t = row & (L_ - 1);
#pragma unroll
  for (int i = 0; i < 6; ++i) {
    int c = lane + 32 * i;
    float y = (v[i] - mu) * rs * g[c] + bta[c];
    int h = c >> 4, f = c & 15;
    Outh[(((size_t)(b * H_ + h)) * L_ + t) * FD_ + f] = (__bf16)(0.5f * y);
  }
}

// ---------------------------------------------------------------------------
// V head-split + transpose to (B*H, DV, L) bf16.
// ---------------------------------------------------------------------------
__global__ void k_v_split(const float* __restrict__ V, __bf16* __restrict__ Vt) {
  long idx = (long)blockIdx.x * blockDim.x + threadIdx.x;
  if (idx >= (long)B_ * H_ * DV_ * L_) return;
  int t   = (int)(idx & (L_ - 1));
  long r  = idx >> 11;
  int dv  = (int)(r & (DV_ - 1));
  int bh  = (int)(r >> 7);
  int b = bh / H_, h = bh % H_;
  Vt[idx] = (__bf16)V[((size_t)(b * L_ + t)) * DM_ + h * DV_ + dv];
}

// ---------------------------------------------------------------------------
// ReBased chunked linear attention. One block per (b,h); 8 waves; sequential
// scan over 16 chunks of 128 tokens. Taylor features synthesized per-lane.
// Output written as swizzled bf16 rows ready for the Wo GEMM.
// LDS ~= 210 KB of the 320 KB WGP budget.
// ---------------------------------------------------------------------------
__global__ __launch_bounds__(256) void k_rebased_attn(
    const __bf16* __restrict__ Qh, const __bf16* __restrict__ Kh,
    const __bf16* __restrict__ Vt, __bf16* __restrict__ O) {
  extern __shared__ char smem[];
  float*  S   = (float*)smem;                                 // [DV_][D2_] S^T
  __bf16* vT  = (__bf16*)(smem + (size_t)DV_ * D2_ * 4);      // [DV_][CH_]
  __bf16* Am  = (__bf16*)((char*)vT + (size_t)DV_ * CH_ * 2); // [CH_][CH_]
  __bf16* qb  = (__bf16*)((char*)Am + (size_t)CH_ * CH_ * 2); // [CH_][FD_]
  __bf16* kb  = (__bf16*)((char*)qb + (size_t)CH_ * FD_ * 2); // [CH_][FD_]
  __bf16* kbT = (__bf16*)((char*)kb + (size_t)CH_ * FD_ * 2); // [FD_][CH_]
  float*  z   = (float*)((char*)kbT + (size_t)FD_ * CH_ * 2); // [D2_]
  float*  den = (float*)((char*)z + (size_t)D2_ * 4);         // [CH_]

  const int tid  = threadIdx.x;
  const int lane = tid & 31, wv = tid >> 5;
  const int half = lane >> 4, l16 = lane & 15;
  const int bh = blockIdx.x;
  const int b = bh / H_, h = bh % H_;

  for (int i = tid; i < DV_ * D2_; i += 256) S[i] = 0.f;
  if (tid < D2_) z[tid] = 0.f;
  __syncthreads();

  const __bf16* Qb = Qh + (size_t)bh * L_ * FD_;
  const __bf16* Kb = Kh + (size_t)bh * L_ * FD_;
  const __bf16* Vb = Vt + (size_t)bh * DV_ * L_;

  for (int c = 0; c < NC_; ++c) {
    // ---- stage q, k, v tiles into LDS (async-to-LDS when available) ----
    {
      const char* qg = (const char*)(Qb + (size_t)c * CH_ * FD_);
      const char* kg = (const char*)(Kb + (size_t)c * CH_ * FD_);
      copy16_g2l((char*)qb + tid * 16, qg + tid * 16);   // 4 KB
      copy16_g2l((char*)kb + tid * 16, kg + tid * 16);   // 4 KB
#pragma unroll
      for (int i = 0; i < 8; ++i) {                      // 32 KB vT
        int idx = tid + i * 256;                         // 0..2047 segments
        int dv  = idx >> 4;
        int sg  = idx & 15;
        copy16_g2l((char*)vT + (size_t)idx * 16,
                   (const char*)(Vb + (size_t)dv * L_ + c * CH_) + sg * 16);
      }
      wait_async_lds();
    }
    __syncthreads();
    for (int i = tid; i < FD_ * CH_; i += 256) {
      int f = i >> 7, tt = i & (CH_ - 1);
      kbT[i] = kb[tt * FD_ + f];
    }
    __syncthreads();

    // per-lane q row (row = wv*16 + l16), reused by A-GEMM and q2@S
    float qr[FD_];
    {
      const __bf16* qp = qb + (wv * 16 + l16) * FD_;
#pragma unroll
      for (int e = 0; e < FD_; ++e) qr[e] = (float)qp[e];
    }

    // ---- A = q2 * k2^T (contraction over D2=256), causal-masked ----
    for (int ct = 0; ct < 8; ++ct) {
      if (ct > wv) {  // fully masked tile (wave-uniform branch)
#pragma unroll
        for (int r = 0; r < 8; ++r)
          Am[(wv * 16 + half * 8 + r) * CH_ + ct * 16 + l16] = (__bf16)0.f;
        continue;
      }
      float kr[FD_];
      const __bf16* kp = kb + (ct * 16 + l16) * FD_;
#pragma unroll
      for (int e = 0; e < FD_; ++e) kr[e] = (float)kp[e];
      v8f acc = zero_v8f();
      for (int ks = 0; ks < D2_; ks += 32) {
        v16bf af, bfv;
#pragma unroll
        for (int e = 0; e < 16; ++e) {
          int da = ks + ((e < 8) ? (half * 8 + e) : (16 + half * 8 + (e - 8)));
          af[e] = (__bf16)(qr[da >> 4] * qr[da & 15]);
          int db = ks + ((e < 8) ? (half * 8 + e) : (16 + half * 8 + (e - 8)));
          bfv[e] = (__bf16)(kr[db >> 4] * kr[db & 15]);
        }
        WMMA_BF16(af, bfv, acc);
      }
#pragma unroll
      for (int r = 0; r < 8; ++r) {
        int i = wv * 16 + half * 8 + r;
        int j = ct * 16 + l16;
        Am[i * CH_ + j] = (__bf16)((j <= i) ? acc[r] : 0.f);
      }
    }
    __syncthreads();

    // ---- den[i] = eps + rowsum(A) + q2[i] . z ----
    if (tid < CH_) {
      int i = tid;
      float s = 1e-12f;
      const __bf16* ar = Am + i * CH_;
      for (int j = 0; j < CH_; ++j) s += (float)ar[j];
      const __bf16* qp = qb + i * FD_;
      float qv[FD_];
#pragma unroll
      for (int f = 0; f < FD_; ++f) qv[f] = (float)qp[f];
      for (int d = 0; d < D2_; ++d) s += qv[d >> 4] * qv[d & 15] * z[d];
      den[i] = s;
    }
    __syncthreads();

    // ---- num = A @ v  +  q2 @ S ; o = num / den (swizzled bf16 out) ----
    __bf16* Orow = O + ((size_t)(b * L_ + c * CH_)) * DM_;
    for (int vt = 0; vt < 8; ++vt) {
      v8f acc = zero_v8f();
      // part 1: A @ v (K = 128 tokens)
      for (int ks = 0; ks < CH_; ks += 32) {
        const __bf16* ap = Am + (wv * 16 + l16) * CH_ + ks + half * 16;
        v8bf a0 = *(const v8bf*)ap;
        v8bf a1 = *(const v8bf*)(ap + 8);
        v16bf af;
#pragma unroll
        for (int e = 0; e < 8; ++e) { af[e] = a0[e]; af[8 + e] = a1[e]; }
        v16bf bfv = *(const v16bf*)(vT + (vt * 16 + l16) * CH_ + ks + half * 16);
        WMMA_BF16(af, bfv, acc);
      }
      // part 2: q2 @ S (K = 256 features; S^T rows contiguous in f32)
      for (int ks = 0; ks < D2_; ks += 32) {
        v16bf af;
#pragma unroll
        for (int e = 0; e < 16; ++e) {
          int da = ks + half * 16 + e;
          af[e] = (__bf16)(qr[da >> 4] * qr[da & 15]);
        }
        const float* sp = S + (size_t)(vt * 16 + l16) * D2_ + ks + half * 16;
        v4f s0 = *(const v4f*)(sp);
        v4f s1 = *(const v4f*)(sp + 4);
        v4f s2 = *(const v4f*)(sp + 8);
        v4f s3 = *(const v4f*)(sp + 12);
        v16bf bfv;
#pragma unroll
        for (int e = 0; e < 4; ++e) {
          bfv[e]      = (__bf16)s0[e];
          bfv[4 + e]  = (__bf16)s1[e];
          bfv[8 + e]  = (__bf16)s2[e];
          bfv[12 + e] = (__bf16)s3[e];
        }
        WMMA_BF16(af, bfv, acc);
      }
      const int scol = swz32(h * DV_ + vt * 16 + l16);
#pragma unroll
      for (int r = 0; r < 8; ++r) {
        int i = wv * 16 + half * 8 + r;
        Orow[(size_t)i * DM_ + scol] = (__bf16)(acc[r] / den[i]);
      }
    }
    __syncthreads();

    // ---- state update: S^T += (k2^T @ v)^T ; z += sum_t k2 ----
    for (int tt = 0; tt < 16; ++tt) {
      int mt = wv * 2 + (tt >> 3);
      int nt = tt & 7;
      int m  = mt * 16 + l16;     // feature row index of k2^T
      int d1 = m >> 4, d2i = m & 15;
      v8f acc = zero_v8f();
      for (int ks = 0; ks < CH_; ks += 32) {
        const __bf16* p1 = kbT + d1 * CH_ + ks + half * 16;
        const __bf16* p2 = kbT + d2i * CH_ + ks + half * 16;
        v8bf x0 = *(const v8bf*)p1,        y0 = *(const v8bf*)p2;
        v8bf x1 = *(const v8bf*)(p1 + 8), y1 = *(const v8bf*)(p2 + 8);
        v16bf af;
#pragma unroll
        for (int e = 0; e < 8; ++e) {
          af[e]     = (__bf16)((float)x0[e] * (float)y0[e]);
          af[8 + e] = (__bf16)((float)x1[e] * (float)y1[e]);
        }
        v16bf bfv = *(const v16bf*)(vT + (nt * 16 + l16) * CH_ + ks + half * 16);
        WMMA_BF16(af, bfv, acc);
      }
#pragma unroll
      for (int r = 0; r < 8; ++r) {
        int d2r = mt * 16 + half * 8 + r;
        int vdx = nt * 16 + l16;
        S[(size_t)vdx * D2_ + d2r] += acc[r];
      }
    }
    {
      int d = tid;
      if (d < D2_) {
        int d1 = d >> 4, d2i = d & 15;
        const __bf16* p1 = kbT + d1 * CH_;
        const __bf16* p2 = kbT + d2i * CH_;
        float s = 0.f;
        for (int t2 = 0; t2 < CH_; ++t2) s += (float)p1[t2] * (float)p2[t2];
        z[d] += s;
      }
    }
    __syncthreads();
  }
}

// ---------------------------------------------------------------------------
extern "C" void kernel_launch(void* const* d_in, const int* in_sizes, int n_in,
                              void* d_out, int out_size, void* d_ws, size_t ws_size,
                              hipStream_t stream) {
  (void)in_sizes; (void)n_in; (void)out_size; (void)ws_size;
  const float* hs = (const float*)d_in[0];
  const float* Wq = (const float*)d_in[1];
  const float* Wk = (const float*)d_in[2];
  const float* Wv = (const float*)d_in[3];
  const float* Wo = (const float*)d_in[4];
  const float* gq = (const float*)d_in[5];
  const float* bq = (const float*)d_in[6];
  const float* gk = (const float*)d_in[7];
  const float* bk = (const float*)d_in[8];
  float* out = (float*)d_out;

  char* ws = (char*)d_ws;
  size_t off = 0;
  auto carve = [&](size_t bytes) -> char* {
    char* p = ws + off;
    off += (bytes + 255) & ~(size_t)255;
    return p;
  };
  __bf16* WqT = (__bf16*)carve((size_t)(H_ * FD_) * DM_ * 2);
  __bf16* WkT = (__bf16*)carve((size_t)(H_ * FD_) * DM_ * 2);
  __bf16* WvT = (__bf16*)carve((size_t)DM_ * DM_ * 2);
  __bf16* WoT = (__bf16*)carve((size_t)DM_ * DM_ * 2);
  __bf16* Xbf = (__bf16*)carve((size_t)BL_ * DM_ * 2);   // swizzled bf16 hs
  float*  Qf  = (float*) carve((size_t)BL_ * (H_ * FD_) * 4);
  float*  Kf  = (float*) carve((size_t)BL_ * (H_ * FD_) * 4);
  float*  Vf  = (float*) carve((size_t)BL_ * DM_ * 4);   // reused for attn out
  __bf16* Qh  = (__bf16*)carve((size_t)B_ * H_ * L_ * FD_ * 2);
  __bf16* Kh  = (__bf16*)carve((size_t)B_ * H_ * L_ * FD_ * 2);
  __bf16* Vt  = (__bf16*)carve((size_t)B_ * H_ * DV_ * L_ * 2);
  __bf16* Obf = (__bf16*)Vf;   // alias: Vf is dead once Vt is built

  // 1) weight transposes (f32 -> bf16, N x K) + hs convert/swizzle
  {
    long tq = (long)DM_ * (H_ * FD_);
    k_wt_bf16<<<dim3((unsigned)((tq + 255) / 256)), dim3(256), 0, stream>>>(Wq, WqT, DM_, H_ * FD_);
    k_wt_bf16<<<dim3((unsigned)((tq + 255) / 256)), dim3(256), 0, stream>>>(Wk, WkT, DM_, H_ * FD_);
    long tv = (long)DM_ * DM_;
    k_wt_bf16<<<dim3((unsigned)((tv + 255) / 256)), dim3(256), 0, stream>>>(Wv, WvT, DM_, DM_);
    k_wt_bf16<<<dim3((unsigned)((tv + 255) / 256)), dim3(256), 0, stream>>>(Wo, WoT, DM_, DM_);
    long tx = (long)BL_ * DM_;
    k_a_swz_bf16<<<dim3((unsigned)((tx + 255) / 256)), dim3(256), 0, stream>>>(hs, Xbf, tx, DM_);
  }
  // 2) projections (WMMA GEMM, pure-bf16 inner loop)
  k_gemm<<<dim3(BL_ / 128, (H_ * FD_) / 64), dim3(256), 0, stream>>>(Xbf, WqT, Qf, BL_, H_ * FD_, DM_);
  k_gemm<<<dim3(BL_ / 128, (H_ * FD_) / 64), dim3(256), 0, stream>>>(Xbf, WkT, Kf, BL_, H_ * FD_, DM_);
  k_gemm<<<dim3(BL_ / 128, DM_ / 64), dim3(256), 0, stream>>>(Xbf, WvT, Vf, BL_, DM_, DM_);
  // 3) layernorm + head split (+0.5 Taylor fold)
  k_ln_split<<<dim3(BL_ / 8), dim3(256), 0, stream>>>(Qf, gq, bq, Qh);
  k_ln_split<<<dim3(BL_ / 8), dim3(256), 0, stream>>>(Kf, gk, bk, Kh);
  // 4) V head-split transpose
  {
    long tv = (long)B_ * H_ * DV_ * L_;
    k_v_split<<<dim3((unsigned)((tv + 255) / 256)), dim3(256), 0, stream>>>(Vf, Vt);
  }
  // 5) chunked linear attention (writes swizzled bf16 into Vf region)
  size_t smem_bytes = (size_t)DV_ * D2_ * 4 + (size_t)DV_ * CH_ * 2 +
                      (size_t)CH_ * CH_ * 2 + (size_t)CH_ * FD_ * 2 * 2 +
                      (size_t)FD_ * CH_ * 2 + (size_t)D2_ * 4 + (size_t)CH_ * 4;
  k_rebased_attn<<<dim3(B_ * H_), dim3(256), smem_bytes, stream>>>(Qh, Kh, Vt, Obf);
  // 6) output projection -> d_out
  k_gemm<<<dim3(BL_ / 128, DM_ / 64), dim3(256), 0, stream>>>(Obf, WoT, out, BL_, DM_, DM_);
}
